// TransformerBlock_50044958933080
// MI455X (gfx1250) — compile-verified, matrix operand order checked by reading, not run on hardware
//
#include <hip/hip_runtime.h>
#include <cmath>

// ---------------------------------------------------------------------------
// Problem dimensions (compile-time constants from the reference)
// ---------------------------------------------------------------------------
static const int Bt = 2, Lt = 192, Dt = 512, Ft = 2048, HNt = 16;
static const int Et = 128, HEt = 4, At = 32, DHt = 32;      // DH = D/HN
static const int HEA = HEt * At;                            // 128

typedef __attribute__((ext_vector_type(16))) __bf16 v16bf;
typedef __attribute__((ext_vector_type(8)))  float  v8f;

union Frag { v16bf v; unsigned int u[8]; };

__device__ __forceinline__ unsigned int pk2(float a, float b) {
  union { __bf16 h[2]; unsigned int u; } p;
  p.h[0] = (__bf16)a; p.h[1] = (__bf16)b;
  return p.u;
}

// ---------------------------------------------------------------------------
// Cooperative load of one 64x32 half-tile (rows x k) of f32 into bf16 LDS.
//   kmajor=0: element (r,k) at P[r*ld + k]   (k contiguous -> vector fast path)
//   kmajor=1: element (r,k) at P[k*ld + r]   (scalar fallback)
// LDS row stride 72 elements (144B): 16B-aligned vector stores, conflict-free
// fragment reads.
// ---------------------------------------------------------------------------
__device__ __forceinline__ void load_tile_half(
    const float* __restrict__ P, int ld, int rows, int r0,
    int K, int kk0, int kmajor, __bf16 (*lds)[72], int ldscol)
{
  const int tid = threadIdx.x;
  if (!kmajor && (r0 + 64 <= rows) && (kk0 + 32 <= K)) {
    int r = tid >> 1, c0 = (tid & 1) << 4;
    const float* src = P + (long)(r0 + r) * ld + kk0 + c0;
    float4 f0 = ((const float4*)src)[0];
    float4 f1 = ((const float4*)src)[1];
    float4 f2 = ((const float4*)src)[2];
    float4 f3 = ((const float4*)src)[3];
    if (kk0 + 64 < K) __builtin_prefetch(src + 64, 0, 1);   // global_prefetch_b8
    uint4 q0, q1;
    q0.x = pk2(f0.x, f0.y); q0.y = pk2(f0.z, f0.w);
    q0.z = pk2(f1.x, f1.y); q0.w = pk2(f1.z, f1.w);
    q1.x = pk2(f2.x, f2.y); q1.y = pk2(f2.z, f2.w);
    q1.z = pk2(f3.x, f3.y); q1.w = pk2(f3.z, f3.w);
    uint4* dst = (uint4*)&lds[r][ldscol + c0];
    dst[0] = q0;
    dst[1] = q1;
  } else {
    for (int idx = tid; idx < 64 * 32; idx += 128) {
      int r = idx >> 5, c = idx & 31;
      int gr = r0 + r, gk = kk0 + c;
      float v = 0.0f;
      if (gr < rows && gk < K)
        v = kmajor ? P[(long)gk * ld + gr] : P[(long)gr * ld + gk];
      lds[r][ldscol + c] = (__bf16)v;
    }
  }
}

// ---------------------------------------------------------------------------
// Generic batched WMMA GEMM:  C[b] = epi(alpha * opA(A[b]) * opB(B[b]) + bias)
// Block tile 64x128, 128 threads (4 waves, 2x2 wave grid), wave tile 32x64
// (2x4 frags of 16x16) -> A fragments reused across 4 B fragments;
// K staged 64 at a time: 16 v_wmma_f32_16x16x32_bf16 per barrier pair.
// Batch index z decomposes as (b0, b1) with independent strides per operand.
// ---------------------------------------------------------------------------
__global__ __launch_bounds__(128) void k_gemm(
    const float* __restrict__ A, const float* __restrict__ B,
    const float* __restrict__ bias, float* __restrict__ C,
    int M, int N, int K, int lda, int ldb, int ldc,
    long sa0, long sa1, long sb0, long sb1, long sc0, long sc1,
    int nb1, int tA, int tB, int accum, int epi, float alpha)
{
  __shared__ __bf16 As[64][72];
  __shared__ __bf16 Bs[128][72];   // stored [n][k]

  const int bz = blockIdx.z;
  const int b0 = bz / nb1, b1 = bz - b0 * nb1;
  A += b0 * sa0 + b1 * sa1;
  B += b0 * sb0 + b1 * sb1;
  C += b0 * sc0 + b1 * sc1;

  const int m0 = blockIdx.y * 64;
  const int n0 = blockIdx.x * 128;
  const int tid = threadIdx.x;
  const int wave = tid >> 5, lane = tid & 31;
  const int wm = (wave >> 1) << 5;     // 0 or 32
  const int wn = (wave & 1) << 6;      // 0 or 64
  const int lhalf = lane >> 4, lr = lane & 15;
  const int kmajA = tA ? 1 : 0;
  const int kmajB = tB ? 0 : 1;

  v8f acc[2][4] = {};

  for (int k0 = 0; k0 < K; k0 += 64) {
    load_tile_half(A, lda, M, m0,      K, k0,      kmajA, As, 0);
    load_tile_half(A, lda, M, m0,      K, k0 + 32, kmajA, As, 32);
    load_tile_half(B, ldb, N, n0,      K, k0,      kmajB, Bs, 0);
    load_tile_half(B, ldb, N, n0,      K, k0 + 32, kmajB, Bs, 32);
    load_tile_half(B, ldb, N, n0 + 64, K, k0,      kmajB, Bs + 64, 0);
    load_tile_half(B, ldb, N, n0 + 64, K, k0 + 32, kmajB, Bs + 64, 32);
    __syncthreads();

#pragma unroll
    for (int ks = 0; ks < 2; ++ks) {
      Frag fa[2], fb[4];
#pragma unroll
      for (int f = 0; f < 2; ++f) {
        int m = wm + (f << 4) + lr;
#pragma unroll
        for (int v = 0; v < 8; ++v) {
          // CDNA5 16-bit A 16x32 layout: VGPR v holds a K pair; lane half
          // selects the K octet within each 16-wide K group.
          int kk = (ks << 5) + ((v >> 2) << 4) + ((v & 3) << 1) + (lhalf << 3);
          fa[f].u[v] = *(const unsigned int*)&As[m][kk];
        }
      }
#pragma unroll
      for (int f = 0; f < 4; ++f) {
        int n = wn + (f << 4) + lr;
#pragma unroll
        for (int v = 0; v < 8; ++v) {
          int kk = (ks << 5) + ((v >> 2) << 4) + ((v & 3) << 1) + (lhalf << 3);
          fb[f].u[v] = *(const unsigned int*)&Bs[n][kk];
        }
      }
#pragma unroll
      for (int fm = 0; fm < 2; ++fm)
#pragma unroll
        for (int fn = 0; fn < 4; ++fn)
          acc[fm][fn] = __builtin_amdgcn_wmma_f32_16x16x32_bf16(
              false, fa[fm].v, false, fb[fn].v, (short)0, acc[fm][fn], false, false);
    }
    __syncthreads();
  }

#pragma unroll
  for (int fm = 0; fm < 2; ++fm)
#pragma unroll
    for (int fn = 0; fn < 4; ++fn) {
      int gn = n0 + wn + (fn << 4) + lr;
#pragma unroll
      for (int v = 0; v < 8; ++v) {
        int gm = m0 + wm + (fm << 4) + v + (lhalf << 3);
        if (gm < M && gn < N) {
          float val = acc[fm][fn][v] * alpha;
          if (bias) val += bias[gn];
          if (epi == 1) val = fmaxf(val, 0.0f);
          else if (epi == 2) val = 1.0f / (1.0f + __expf(-val));
          long off = (long)gm * ldc + gn;
          if (accum) val += C[off];
          C[off] = val;
        }
      }
    }
}

// ---------------------------------------------------------------------------
// Wave-per-row LayerNorm (wave32)
// ---------------------------------------------------------------------------
__global__ __launch_bounds__(128) void k_layernorm(
    const float* __restrict__ x, const float* __restrict__ g,
    const float* __restrict__ b, float* __restrict__ y, int R, int C)
{
  int row = blockIdx.x * 4 + (threadIdx.x >> 5);
  int lane = threadIdx.x & 31;
  if (row >= R) return;
  const float* xr = x + (long)row * C;
  float s = 0.0f, s2 = 0.0f;
  for (int c = lane; c < C; c += 32) { float v = xr[c]; s += v; s2 += v * v; }
  for (int o = 16; o; o >>= 1) { s += __shfl_xor(s, o, 32); s2 += __shfl_xor(s2, o, 32); }
  float mu = s / C;
  float var = s2 / C - mu * mu;
  float r = rsqrtf(var + 1e-5f);
  float* yr = y + (long)row * C;
  for (int c = lane; c < C; c += 32) yr[c] = (xr[c] - mu) * r * g[c] + b[c];
}

// ---------------------------------------------------------------------------
// Wave-per-row softmax (in place), handles all -inf rows
// ---------------------------------------------------------------------------
__global__ __launch_bounds__(128) void k_softmax(float* __restrict__ x, int R, int C)
{
  int row = blockIdx.x * 4 + (threadIdx.x >> 5);
  int lane = threadIdx.x & 31;
  if (row >= R) return;
  float* xr = x + (long)row * C;
  float mx = -INFINITY;
  for (int c = lane; c < C; c += 32) mx = fmaxf(mx, xr[c]);
  for (int o = 16; o; o >>= 1) mx = fmaxf(mx, __shfl_xor(mx, o, 32));
  if (mx == -INFINITY) {
    for (int c = lane; c < C; c += 32) xr[c] = 0.0f;
    return;
  }
  float s = 0.0f;
  for (int c = lane; c < C; c += 32) { float e = __expf(xr[c] - mx); xr[c] = e; s += e; }
  for (int o = 16; o; o >>= 1) s += __shfl_xor(s, o, 32);
  float inv = 1.0f / s;
  for (int c = lane; c < C; c += 32) xr[c] *= inv;
}

// ---------------------------------------------------------------------------
// Elementwise helpers
// ---------------------------------------------------------------------------
__global__ void k_mask_node(const float* __restrict__ x, const unsigned char* __restrict__ pm,
                            float* __restrict__ y, int n)
{
  int i = blockIdx.x * 256 + threadIdx.x;
  if (i >= n) return;
  int row = i / Dt;
  y[i] = pm[row] ? 0.0f : x[i];
}

// qk[B,HN,L,L] += edge bias (with key mask); also emit aw[B,L,L,HN]
__global__ void k_logits_fuse(float* __restrict__ qk, const float* __restrict__ eb,
                              const unsigned char* __restrict__ pm,
                              float* __restrict__ aw, int n)
{
  int i = blockIdx.x * 256 + threadIdx.x;
  if (i >= n) return;
  int j = i % Lt;
  int q = (i / Lt) % Lt;
  int h = (i / (Lt * Lt)) % HNt;
  int b = i / (HNt * Lt * Lt);
  long eidx = (((long)(b * Lt + q) * Lt) + j) * HNt + h;
  float v, a;
  if (pm[b * Lt + j]) { v = -INFINITY; a = 0.0f; }
  else { v = qk[i] + eb[eidx]; a = v; }
  qk[i] = v;
  aw[eidx] = a;
}

__global__ void k_add(const float* __restrict__ a, const float* __restrict__ b,
                      float* __restrict__ o, int n)
{
  int i = blockIdx.x * 256 + threadIdx.x;
  if (i < n) o[i] = a[i] + b[i];
}

__global__ void k_mul(float* __restrict__ o, const float* __restrict__ a,
                      const float* __restrict__ b, int n)
{
  int i = blockIdx.x * 256 + threadIdx.x;
  if (i < n) o[i] = a[i] * b[i];
}

__global__ void k_fma_res(float* __restrict__ acc_, const float* __restrict__ g,
                          const float* __restrict__ t, int n)
{
  int i = blockIdx.x * 256 + threadIdx.x;
  if (i < n) acc_[i] += g[i] * t[i];
}

// a = emask * sigmoid(gate) * proj   (gate already sigmoid'ed by GEMM epilogue)
__global__ void k_tu_gate(const float* __restrict__ gs, const float* __restrict__ pp,
                          const unsigned char* __restrict__ pm, float* __restrict__ o, int n)
{
  int i = blockIdx.x * 256 + threadIdx.x;
  if (i >= n) return;
  int j = (i / Et) % Lt;
  int r = (i / (Et * Lt)) % Lt;
  int b = i / (Et * Lt * Lt);
  float m = (!pm[b * Lt + r] && !pm[b * Lt + j]) ? 1.0f : 0.0f;
  o[i] = m * gs[i] * pp[i];
}

// [B,L,L,E] -> [B,E,L,L]
__global__ void k_perm_to_cf(const float* __restrict__ in, float* __restrict__ out, int n)
{
  int i = blockIdx.x * 256 + threadIdx.x;
  if (i >= n) return;
  int c = i % Et;
  int j = (i / Et) % Lt;
  int r = (i / (Et * Lt)) % Lt;
  int b = i / (Et * Lt * Lt);
  out[(((long)b * Et + c) * Lt + r) * Lt + j] = in[i];
}

// [B,E,L,L] -> [B,L,L,E]
__global__ void k_perm_from_cf(const float* __restrict__ in, float* __restrict__ out, int n)
{
  int i = blockIdx.x * 256 + threadIdx.x;
  if (i >= n) return;
  int j = i % Lt;
  int r = (i / Lt) % Lt;
  int c = (i / (Lt * Lt)) % Et;
  int b = i / (Et * Lt * Lt);
  out[(((long)b * Lt + r) * Lt + j) * Et + c] = in[i];
}

// lg chunk [CH,HE,L,L] += -1e9 where (i or k) padded
__global__ void k_add_mbias(float* __restrict__ lg, const unsigned char* __restrict__ pm,
                            int b, int i0, int n)
{
  int idx = blockIdx.x * 256 + threadIdx.x;
  if (idx >= n) return;
  int kk = idx % Lt;
  int il = idx / (HEt * Lt * Lt);
  int i = i0 + il;
  if (pm[b * Lt + i] || pm[b * Lt + kk]) lg[idx] += -1e9f;
}

// ---------------------------------------------------------------------------
// Host-side launcher
// ---------------------------------------------------------------------------
static inline void launch_gemm(hipStream_t s,
    const float* A, const float* B, const float* bias, float* C,
    int M, int N, int K, int lda, int ldb, int ldc,
    long sa0, long sa1, long sb0, long sb1, long sc0, long sc1,
    int nb0, int nb1, int tA, int tB, int accum, int epi, float alpha)
{
  dim3 g((N + 127) / 128, (M + 63) / 64, nb0 * nb1);
  k_gemm<<<g, 128, 0, s>>>(A, B, bias, C, M, N, K, lda, ldb, ldc,
                           sa0, sa1, sb0, sb1, sc0, sc1,
                           nb1, tA, tB, accum, epi, alpha);
}

static inline dim3 ew(int n) { return dim3((n + 255) / 256); }

extern "C" void kernel_launch(void* const* d_in, const int* in_sizes, int n_in,
                              void* d_out, int out_size, void* d_ws, size_t ws_size,
                              hipStream_t stream)
{
  (void)in_sizes; (void)n_in; (void)out_size; (void)ws_size;

  const float* node = (const float*)d_in[0];                 // [B,L,D]
  const float* edge = (const float*)d_in[1];                 // [B,L,L,E]
  const unsigned char* pm = (const unsigned char*)d_in[2];   // [B,L] bool

  // params in dict insertion order
  const float* eb_w   = (const float*)d_in[3];
  const float* eb_b   = (const float*)d_in[4];
  const float* ln1_g  = (const float*)d_in[5];
  const float* ln1_b  = (const float*)d_in[6];
  const float* in_w   = (const float*)d_in[7];
  const float* in_b   = (const float*)d_in[8];
  const float* out_w  = (const float*)d_in[9];
  const float* out_b  = (const float*)d_in[10];
  const float* ln2_g  = (const float*)d_in[11];
  const float* ln2_b  = (const float*)d_in[12];
  const float* fc1_w  = (const float*)d_in[13];
  const float* fc1_b  = (const float*)d_in[14];
  const float* fc2_w  = (const float*)d_in[15];
  const float* fc2_b  = (const float*)d_in[16];
  const float* n2e_w1 = (const float*)d_in[17];
  const float* n2e_b1 = (const float*)d_in[18];
  const float* n2e_w2 = (const float*)d_in[19];
  const float* n2e_b2 = (const float*)d_in[20];
  const float* tu_ln_g  = (const float*)d_in[21];
  const float* tu_ln_b  = (const float*)d_in[22];
  const float* tu_ap_w  = (const float*)d_in[23];
  const float* tu_ap_b  = (const float*)d_in[24];
  const float* tu_ag_w  = (const float*)d_in[25];
  const float* tu_ag_b  = (const float*)d_in[26];
  const float* tu_bp_w  = (const float*)d_in[27];
  const float* tu_bp_b  = (const float*)d_in[28];
  const float* tu_bg_w  = (const float*)d_in[29];
  const float* tu_bg_b  = (const float*)d_in[30];
  const float* tu_ln2_g = (const float*)d_in[31];
  const float* tu_ln2_b = (const float*)d_in[32];
  const float* tu_g_w   = (const float*)d_in[33];
  const float* tu_g_b   = (const float*)d_in[34];
  const float* tu_z_w   = (const float*)d_in[35];
  const float* tu_z_b   = (const float*)d_in[36];
  const float* ta_ln_g  = (const float*)d_in[37];
  const float* ta_ln_b  = (const float*)d_in[38];
  const float* ta_q_w   = (const float*)d_in[39];
  const float* ta_k_w   = (const float*)d_in[40];
  const float* ta_v_w   = (const float*)d_in[41];
  const float* ta_g_w   = (const float*)d_in[42];
  const float* ta_g_b   = (const float*)d_in[43];
  const float* ta_o_w   = (const float*)d_in[44];
  const float* ta_o_b   = (const float*)d_in[45];
  const float* tr_ln_g  = (const float*)d_in[46];
  const float* tr_ln_b  = (const float*)d_in[47];
  const float* tr_w1    = (const float*)d_in[48];
  const float* tr_b1    = (const float*)d_in[49];
  const float* tr_w2    = (const float*)d_in[50];
  const float* tr_b2    = (const float*)d_in[51];

  // ---- workspace layout (floats) ----
  const long NE  = (long)Bt * Lt * Lt * Et;     // 9,437,184
  const long BLL = (long)Bt * Lt * Lt;          // 73,728
  const long BLD = (long)Bt * Lt * Dt;          // 196,608
  float* ws = (float*)d_ws;
  float* E1 = ws;                // edge state
  float* E2 = ws + 1 * NE;       // LN(z)
  float* E3 = ws + 2 * NE;
  float* E4 = ws + 3 * NE;
  float* E5 = ws + 4 * NE;
  float* E6 = ws + 5 * NE;
  float* E7 = ws + 6 * NE;
  float* LG = ws + 7 * NE;                        // 2,359,296 (16-i chunk of logits)
  float* S  = ws + 7 * NE + 2359296;
  float* node_m = S;                    // 196608
  float* xln    = S + 196608;           // 196608 (also reused as tmp)
  float* qkv    = S + 393216;           // 589824
  float* logits = S + 983040;           // 1179648  [B,HN,L,L]
  float* ebias  = S + 2162688;          // 1179648  [B,L,L,HN]
  float* aw     = S + 3342336;          // 1179648  [B,L,L,HN]
  float* ocat   = S + 4521984;          // 196608
  float* x1     = S + 4718592;          // 196608
  float* yln    = S + 4915200;          // 196608
  float* hffn   = S + 5111808;          // 786432
  float* n2et   = S + 5898240;          // 1179648

  float* outNode = (float*)d_out;
  float* outEdge = (float*)d_out + BLD;

  const float qs = 0.17677669529663687f;   // 1/sqrt(32) for both attentions

  // ===================== node update =====================
  k_mask_node<<<ew((int)BLD), 256, 0, stream>>>(node, pm, node_m, (int)BLD);
  k_layernorm<<<(Bt * Lt + 3) / 4, 128, 0, stream>>>(node_m, ln1_g, ln1_b, xln, Bt * Lt, Dt);
  // qkv = xln @ in_w + in_b
  launch_gemm(stream, xln, in_w, in_b, qkv, Bt * Lt, 3 * Dt, Dt, Dt, 3 * Dt, 3 * Dt,
              0, 0, 0, 0, 0, 0, 1, 1, 0, 0, 0, 0, 1.0f);
  // ebias = edge @ eb_w + eb_b    [B*L*L, HN]
  launch_gemm(stream, edge, eb_w, eb_b, ebias, (int)BLL, HNt, Et, Et, HNt, HNt,
              0, 0, 0, 0, 0, 0, 1, 1, 0, 0, 0, 0, 1.0f);
  // logits = qs * q @ k^T   (batched over (b,h))
  launch_gemm(stream, qkv, qkv + Dt, nullptr, logits,
              Lt, Lt, DHt, 3 * Dt, 3 * Dt, Lt,
              (long)Lt * 3 * Dt, DHt, (long)Lt * 3 * Dt, DHt,
              (long)HNt * Lt * Lt, (long)Lt * Lt,
              Bt, HNt, 0, 1, 0, 0, qs);
  k_logits_fuse<<<ew(Bt * HNt * Lt * Lt), 256, 0, stream>>>(logits, ebias, pm, aw,
                                                            Bt * HNt * Lt * Lt);
  k_softmax<<<(Bt * HNt * Lt + 3) / 4, 128, 0, stream>>>(logits, Bt * HNt * Lt, Lt);
  // ocat = probs @ v
  launch_gemm(stream, logits, qkv + 2 * Dt, nullptr, ocat,
              Lt, DHt, Lt, Lt, 3 * Dt, Dt,
              (long)HNt * Lt * Lt, (long)Lt * Lt, (long)Lt * 3 * Dt, DHt,
              (long)Lt * Dt, DHt,
              Bt, HNt, 0, 0, 0, 0, 1.0f);
  // x1 = node_m + ocat @ out_w + out_b
  launch_gemm(stream, ocat, out_w, out_b, xln, Bt * Lt, Dt, Dt, Dt, Dt, Dt,
              0, 0, 0, 0, 0, 0, 1, 1, 0, 0, 0, 0, 1.0f);
  k_add<<<ew((int)BLD), 256, 0, stream>>>(node_m, xln, x1, (int)BLD);
  // FFN
  k_layernorm<<<(Bt * Lt + 3) / 4, 128, 0, stream>>>(x1, ln2_g, ln2_b, yln, Bt * Lt, Dt);
  launch_gemm(stream, yln, fc1_w, fc1_b, hffn, Bt * Lt, Ft, Dt, Dt, Ft, Ft,
              0, 0, 0, 0, 0, 0, 1, 1, 0, 0, 0, 1, 1.0f);           // relu
  launch_gemm(stream, hffn, fc2_w, fc2_b, xln, Bt * Lt, Dt, Ft, Ft, Dt, Dt,
              0, 0, 0, 0, 0, 0, 1, 1, 0, 0, 0, 0, 1.0f);
  k_add<<<ew((int)BLD), 256, 0, stream>>>(x1, xln, outNode, (int)BLD);

  // ===================== node2edge =====================
  launch_gemm(stream, aw, n2e_w1, n2e_b1, n2et, (int)BLL, HNt, HNt, HNt, HNt, HNt,
              0, 0, 0, 0, 0, 0, 1, 1, 0, 0, 0, 1, 1.0f);           // relu
  hipMemcpyAsync(E1, edge, NE * sizeof(float), hipMemcpyDeviceToDevice, stream);
  launch_gemm(stream, n2et, n2e_w2, n2e_b2, E1, (int)BLL, Et, HNt, HNt, Et, Et,
              0, 0, 0, 0, 0, 0, 1, 1, 0, 0, 1, 0, 1.0f);           // accum into edge

  // ===================== triangle multiplicative update =====================
  k_layernorm<<<((int)BLL + 3) / 4, 128, 0, stream>>>(E1, tu_ln_g, tu_ln_b, E2, (int)BLL, Et);
  // a = emask * sigmoid(z@ag) * (z@ap)
  launch_gemm(stream, E2, tu_ag_w, tu_ag_b, E3, (int)BLL, Et, Et, Et, Et, Et,
              0, 0, 0, 0, 0, 0, 1, 1, 0, 0, 0, 2, 1.0f);           // sigmoid
  launch_gemm(stream, E2, tu_ap_w, tu_ap_b, E4, (int)BLL, Et, Et, Et, Et, Et,
              0, 0, 0, 0, 0, 0, 1, 1, 0, 0, 0, 0, 1.0f);
  k_tu_gate<<<ew((int)NE), 256, 0, stream>>>(E3, E4, pm, E5, (int)NE);
  // b = emask * sigmoid(z@bg) * (z@bp)
  launch_gemm(stream, E2, tu_bg_w, tu_bg_b, E3, (int)BLL, Et, Et, Et, Et, Et,
              0, 0, 0, 0, 0, 0, 1, 1, 0, 0, 0, 2, 1.0f);
  launch_gemm(stream, E2, tu_bp_w, tu_bp_b, E4, (int)BLL, Et, Et, Et, Et, Et,
              0, 0, 0, 0, 0, 0, 1, 1, 0, 0, 0, 0, 1.0f);
  k_tu_gate<<<ew((int)NE), 256, 0, stream>>>(E3, E4, pm, E6, (int)NE);
  // channels-first, batched 192^3 GEMMs:  ab = a b^T + a^T b  (per b,c)
  k_perm_to_cf<<<ew((int)NE), 256, 0, stream>>>(E5, E7, (int)NE);    // a_cf
  k_perm_to_cf<<<ew((int)NE), 256, 0, stream>>>(E6, E5, (int)NE);    // b_cf
  launch_gemm(stream, E7, E5, nullptr, E6, Lt, Lt, Lt, Lt, Lt, Lt,
              (long)Lt * Lt, 0, (long)Lt * Lt, 0, (long)Lt * Lt, 0,
              Bt * Et, 1, 0, 1, 0, 0, 1.0f);                        // a @ b^T
  launch_gemm(stream, E7, E5, nullptr, E6, Lt, Lt, Lt, Lt, Lt, Lt,
              (long)Lt * Lt, 0, (long)Lt * Lt, 0, (long)Lt * Lt, 0,
              Bt * Et, 1, 1, 0, 1, 0, 1.0f);                        // += a^T @ b
  k_perm_from_cf<<<ew((int)NE), 256, 0, stream>>>(E6, E3, (int)NE);  // ab
  k_layernorm<<<((int)BLL + 3) / 4, 128, 0, stream>>>(E3, tu_ln2_g, tu_ln2_b, E4, (int)BLL, Et);
  launch_gemm(stream, E4, tu_z_w, tu_z_b, E5, (int)BLL, Et, Et, Et, Et, Et,
              0, 0, 0, 0, 0, 0, 1, 1, 0, 0, 0, 0, 1.0f);
  launch_gemm(stream, E2, tu_g_w, tu_g_b, E7, (int)BLL, Et, Et, Et, Et, Et,
              0, 0, 0, 0, 0, 0, 1, 1, 0, 0, 0, 2, 1.0f);            // sigmoid gate
  k_fma_res<<<ew((int)NE), 256, 0, stream>>>(E1, E7, E5, (int)NE);   // edge += g*proj

  // ===================== row-wise gated triangle attention =====================
  k_layernorm<<<((int)BLL + 3) / 4, 128, 0, stream>>>(E1, ta_ln_g, ta_ln_b, E2, (int)BLL, Et);
  launch_gemm(stream, E2, ta_q_w, nullptr, E3, (int)BLL, HEA, Et, Et, HEA, HEA,
              0, 0, 0, 0, 0, 0, 1, 1, 0, 0, 0, 0, 1.0f);
  launch_gemm(stream, E2, ta_k_w, nullptr, E4, (int)BLL, HEA, Et, Et, HEA, HEA,
              0, 0, 0, 0, 0, 0, 1, 1, 0, 0, 0, 0, 1.0f);
  launch_gemm(stream, E2, ta_v_w, nullptr, E5, (int)BLL, HEA, Et, Et, HEA, HEA,
              0, 0, 0, 0, 0, 0, 1, 1, 0, 0, 0, 0, 1.0f);
  launch_gemm(stream, E2, ta_g_w, ta_g_b, E6, (int)BLL, HEA, Et, Et, HEA, HEA,
              0, 0, 0, 0, 0, 0, 1, 1, 0, 0, 0, 2, 1.0f);            // sigmoid gate
  const int CH = 16;                          // i-rows per chunk
  const int nLG = CH * HEt * Lt * Lt;         // 2,359,296
  for (int b = 0; b < Bt; ++b)
    for (int i0 = 0; i0 < Lt; i0 += CH) {
      long boff = ((long)(b * Lt + i0)) * Lt * HEA;
      // lg[il,h,j,k] = qs * q.kT
      launch_gemm(stream, E3 + boff, E4 + boff, nullptr, LG,
                  Lt, Lt, At, HEA, HEA, Lt,
                  (long)Lt * HEA, At, (long)Lt * HEA, At,
                  (long)HEt * Lt * Lt, (long)Lt * Lt,
                  CH, HEt, 0, 1, 0, 0, qs);
      k_add_mbias<<<ew(nLG), 256, 0, stream>>>(LG, pm, b, i0, nLG);
      k_softmax<<<(CH * HEt * Lt + 3) / 4, 128, 0, stream>>>(LG, CH * HEt * Lt, Lt);
      // oa[il,j,h,a] = pr @ v
      launch_gemm(stream, LG, E5 + boff, nullptr, E7 + boff,
                  Lt, At, Lt, Lt, HEA, HEA,
                  (long)HEt * Lt * Lt, (long)Lt * Lt,
                  (long)Lt * HEA, At, (long)Lt * HEA, At,
                  CH, HEt, 0, 0, 0, 0, 1.0f);
    }
  k_mul<<<ew((int)NE), 256, 0, stream>>>(E6, E6, E7, (int)NE);       // gated = g * oa
  launch_gemm(stream, E6, ta_o_w, ta_o_b, E1, (int)BLL, Et, HEA, HEA, Et, Et,
              0, 0, 0, 0, 0, 0, 1, 1, 0, 0, 1, 0, 1.0f);            // edge += out proj

  // ===================== transition (no residual) =====================
  k_layernorm<<<((int)BLL + 3) / 4, 128, 0, stream>>>(E1, tr_ln_g, tr_ln_b, E2, (int)BLL, Et);
  launch_gemm(stream, E2, tr_w1, tr_b1, E3, (int)BLL, 2 * Et, Et, Et, 2 * Et, 2 * Et,
              0, 0, 0, 0, 0, 0, 1, 1, 0, 0, 0, 1, 1.0f);            // relu (spans E3..E4)
  launch_gemm(stream, E3, tr_w2, tr_b2, outEdge, (int)BLL, Et, 2 * Et, 2 * Et, Et, Et,
              0, 0, 0, 0, 0, 0, 1, 1, 0, 0, 0, 0, 1.0f);
}
